// CharRNN_19885698581110
// MI455X (gfx1250) — compile-verified, matrix operand order checked by reading, not run on hardware
//
#include <hip/hip_runtime.h>

#define T_SEQ 1024
#define B_SZ  128
#define V_SZ  256
#define E_SZ  128
#define H_SZ  1024
#define D_SZ  512
#define S_SZ  5
#define G4H   4096          // 4*H
#define XPAD  160           // padded xin width (E + S = 133 -> 160)
#define KTOT  1184          // H + XPAD = 37 * 32
#define XUSED 133
#define CH    128           // K staging chunk (elements)
#define CHP   (CH + 8)      // LDS row stride (bank-conflict padding)

typedef __attribute__((ext_vector_type(16))) __bf16 v16bf;
typedef __attribute__((ext_vector_type(8)))  float  v8f;
typedef int v4i_t __attribute__((vector_size(16)));
typedef __attribute__((address_space(1))) v4i_t* as1_v4i;
typedef __attribute__((address_space(3))) v4i_t* as3_v4i;

// CDNA5 async global->LDS path (availability differs across toolchains)
#if __has_builtin(__builtin_amdgcn_global_load_async_to_lds_b128) && \
    __has_builtin(__builtin_amdgcn_s_wait_asynccnt)
#define HAVE_ASYNC_LDS 1
#endif

// ---------- helpers ----------
__device__ __forceinline__ unsigned short f2bf(float f) {
  unsigned u = __float_as_uint(f);
  u = (u + 0x7FFFu + ((u >> 16) & 1u)) >> 16;   // round-to-nearest-even bf16
  return (unsigned short)u;
}

// Load one 16-bf16 WMMA operand fragment for a lane.
// 16-bit A/B layout (ISA 7.12.2): lane covers row (lane%16); half = lane/16
// selects K chunks {8*half..8*half+7} and {16+8*half..23+8*half}.
__device__ __forceinline__ v16bf load16(const unsigned short* __restrict__ row,
                                        int k0, int half) {
  union { uint4 q[2]; v16bf v; } u;
  u.q[0] = *(const uint4*)(row + k0 + (half << 3));
  u.q[1] = *(const uint4*)(row + k0 + 16 + (half << 3));
  return u.v;
}

// 16-byte global -> LDS staging (async on gfx1250 when available)
__device__ __forceinline__ void stage16(unsigned short* dst,
                                        const unsigned short* src) {
#ifdef HAVE_ASYNC_LDS
  __builtin_amdgcn_global_load_async_to_lds_b128(
      (as1_v4i)(v4i_t*)(uintptr_t)src,   // global source (AS1 v4i*)
      (as3_v4i)(v4i_t*)dst,              // LDS destination (AS3 v4i*)
      0, 0);
#else
  *(uint4*)dst = *(const uint4*)src;
#endif
}

__device__ __forceinline__ void stage_fence() {
#ifdef HAVE_ASYNC_LDS
  __builtin_amdgcn_s_wait_asynccnt(0);
#endif
  __syncthreads();
}

__device__ __forceinline__ float sigmf(float x) {
  return 1.0f / (1.0f + __expf(-x));
}

// ---------- prep kernels ----------

// Pack fused recurrent weight W' [4096 x 1184] bf16:
// cols [0,1024): Whh ; cols [1024,1157): Wih ; rest zero.
__global__ void pack_wp_kernel(const float* __restrict__ Whh,
                               const float* __restrict__ Wih,
                               unsigned short* __restrict__ Wp) {
  const long long N = (long long)G4H * KTOT;
  for (long long i = (long long)blockIdx.x * blockDim.x + threadIdx.x;
       i < N; i += (long long)gridDim.x * blockDim.x) {
    int n = (int)(i / KTOT);
    int k = (int)(i % KTOT);
    float v = 0.0f;
    if (k < H_SZ)                v = Whh[(long long)n * H_SZ + k];
    else if (k < H_SZ + XUSED)   v = Wih[(long long)n * XUSED + (k - H_SZ)];
    Wp[i] = f2bf(v);
  }
}

// Generic f32 -> bf16 cast (Wd, Wo).
__global__ void cast_bf16_kernel(const float* __restrict__ src,
                                 unsigned short* __restrict__ dst, int n) {
  for (int i = blockIdx.x * blockDim.x + threadIdx.x; i < n;
       i += gridDim.x * blockDim.x)
    dst[i] = f2bf(src[i]);
}

// bias = bih + bhh ; zero c-state ; zero h0.
__global__ void prep_misc_kernel(const float* __restrict__ bih,
                                 const float* __restrict__ bhh,
                                 float* __restrict__ bias,
                                 float* __restrict__ cstate,
                                 unsigned short* __restrict__ h0) {
  int i = blockIdx.x * blockDim.x + threadIdx.x;       // 131072 threads
  if (i < G4H) bias[i] = bih[i] + bhh[i];
  if (i < B_SZ * H_SZ) { cstate[i] = 0.0f; h0[i] = 0; }
}

// ---------- embedding gather + senti blocks (pointwise in time) ----------
__global__ void embed_senti_kernel(const int* __restrict__ x,
                                   const float* __restrict__ emb,
                                   const float* __restrict__ s1_Wih,
                                   const float* __restrict__ s1_bih,
                                   const float* __restrict__ s1_bhh,
                                   const float* __restrict__ s1_Wd,
                                   const float* __restrict__ s1_bd,
                                   const float* __restrict__ s2_Wih,
                                   const float* __restrict__ s2_bih,
                                   const float* __restrict__ s2_bhh,
                                   const float* __restrict__ s2_Wd,
                                   const float* __restrict__ s2_bd,
                                   unsigned short* __restrict__ xin16) {
  int idx = blockIdx.x * blockDim.x + threadIdx.x;
  if (idx >= T_SEQ * B_SZ) return;
  int t = idx >> 7;            // / 128
  int b = idx & 127;

  int ch = x[(long long)b * T_SEQ + t] & (V_SZ - 1);   // mask for safety
  const float* er = emb + (long long)ch * E_SZ;

  unsigned short* xrow = xin16 + (long long)idx * XPAD;

  // senti block 1: E(128) -> 4*8 gates
  float g1[32];
  #pragma unroll
  for (int j = 0; j < 32; ++j) {
    const float* w = s1_Wih + j * E_SZ;
    float acc = 0.0f;
    for (int k = 0; k < E_SZ; ++k) acc = fmaf(w[k], er[k], acc);
    g1[j] = acc + s1_bih[j] + s1_bhh[j];
  }
  float h1[8];
  #pragma unroll
  for (int u = 0; u < 8; ++u) {
    float c = sigmf(g1[u]) * tanhf(g1[16 + u]);
    h1[u] = sigmf(g1[24 + u]) * tanhf(c);
  }
  float l1[5], a1[5];
  float m1 = -1e30f;
  #pragma unroll
  for (int j = 0; j < 5; ++j) {
    float acc = s1_bd[j];
    #pragma unroll
    for (int u = 0; u < 8; ++u) acc = fmaf(s1_Wd[j * 8 + u], h1[u], acc);
    l1[j] = acc; m1 = fmaxf(m1, acc);
  }
  float z1 = 0.0f;
  #pragma unroll
  for (int j = 0; j < 5; ++j) { a1[j] = __expf(l1[j] - m1); z1 += a1[j]; }
  float rz1 = 1.0f / z1;
  #pragma unroll
  for (int j = 0; j < 5; ++j) a1[j] *= rz1;

  // senti block 2: S(5) -> 4*8 gates
  float g2[32];
  #pragma unroll
  for (int j = 0; j < 32; ++j) {
    const float* w = s2_Wih + j * S_SZ;
    float acc = s2_bih[j] + s2_bhh[j];
    #pragma unroll
    for (int k = 0; k < S_SZ; ++k) acc = fmaf(w[k], a1[k], acc);
    g2[j] = acc;
  }
  float h2[8];
  #pragma unroll
  for (int u = 0; u < 8; ++u) {
    float c = sigmf(g2[u]) * tanhf(g2[16 + u]);
    h2[u] = sigmf(g2[24 + u]) * tanhf(c);
  }
  float l2[5], a2[5];
  float m2 = -1e30f;
  #pragma unroll
  for (int j = 0; j < 5; ++j) {
    float acc = s2_bd[j];
    #pragma unroll
    for (int u = 0; u < 8; ++u) acc = fmaf(s2_Wd[j * 8 + u], h2[u], acc);
    l2[j] = acc; m2 = fmaxf(m2, acc);
  }
  float z2 = 0.0f;
  #pragma unroll
  for (int j = 0; j < 5; ++j) { a2[j] = __expf(l2[j] - m2); z2 += a2[j]; }
  float rz2 = 1.0f / z2;
  #pragma unroll
  for (int j = 0; j < 5; ++j) a2[j] *= rz2;

  // write e_t (bf16) + pad zeros
  for (int k = 0; k < E_SZ; ++k) xrow[k] = f2bf(er[k]);
  for (int k = XUSED; k < XPAD; ++k) xrow[k] = 0;

  // attn scatter: a2_t feeds step t+1 ; step 0 gets zeros
  if (t == 0) {
    for (int j = 0; j < S_SZ; ++j) xrow[E_SZ + j] = 0;
  }
  if (t + 1 < T_SEQ) {
    unsigned short* xnext = xin16 + ((long long)(t + 1) * B_SZ + b) * XPAD;
    for (int j = 0; j < S_SZ; ++j) xnext[E_SZ + j] = f2bf(a2[j]);
  }
}

// ---------- one LSTM time step (LDS-staged): ----------
// g = [h_{t-1}, xin_t] @ W'^T + bias ; fused cell update.
// Grid: 64 blocks (one per 16 h-columns), 512 threads = 16 waves.
// W' is read exactly once per step; A (h_prev) streamed from L2.
// Wave w: m-tile = w/2 (16 batch rows), gate pair = {2*(w&1), 2*(w&1)+1}.
__global__ __launch_bounds__(512)
void lstm_step_kernel(const unsigned short* __restrict__ hprev,  // [128,1024] bf16
                      const unsigned short* __restrict__ xin_t,  // [128,160]  bf16
                      const unsigned short* __restrict__ Wp,     // [4096,1184] bf16
                      const float* __restrict__ bias,            // [4096]
                      float* __restrict__ cstate,                // [128,1024]
                      unsigned short* __restrict__ hout) {       // [128,1024] bf16
  __shared__ __align__(16) unsigned short A_s[B_SZ][CHP];   // 34.8 KB
  __shared__ __align__(16) unsigned short B_s[64][CHP];     // 17.4 KB
  __shared__ __align__(16) float gtile[8][4][16][16];       // 32 KB

  const int nt   = blockIdx.x;           // 0..63 (h-column tile)
  const int tid  = threadIdx.x;
  const int wave = tid >> 5;             // 0..15
  const int lane = tid & 31;
  const int half = lane >> 4;
  const int ln   = lane & 15;
  const int mt   = wave >> 1;            // 0..7
  const int g0   = (wave & 1) << 1;      // gates {g0, g0+1}

  v8f acc0 = {0.f, 0.f, 0.f, 0.f, 0.f, 0.f, 0.f, 0.f};
  v8f acc1 = {0.f, 0.f, 0.f, 0.f, 0.f, 0.f, 0.f, 0.f};

  // K loop over 10 chunks: c 0..7 -> h part (8*128), c==8 -> xin[0:128),
  // c==9 -> xin[128:160) (32 wide).
  for (int c = 0; c < 10; ++c) {
    const int len = (c < 9) ? CH : 32;   // chunk width (elements)
    const int lw  = len >> 3;            // 16B groups per row

    // stage A chunk [128 x len]
    for (int q = tid; q < B_SZ * lw; q += 512) {
      int row = q / lw, col = (q - row * lw) << 3;
      const unsigned short* src = (c < 8)
          ? hprev + (long long)row * H_SZ + (c << 7) + col
          : xin_t + (long long)row * XPAD + ((c - 8) << 7) + col;
      stage16(&A_s[row][col], src);
    }
    // stage B chunk [64 x len] : local row r -> W' row (r/16)*1024 + nt*16 + r%16
    for (int q = tid; q < 64 * lw; q += 512) {
      int row = q / lw, col = (q - row * lw) << 3;
      const unsigned short* src =
          Wp + (long long)((row >> 4) * H_SZ + nt * 16 + (row & 15)) * KTOT +
          (c << 7) + col;
      __builtin_prefetch(src + CH, 0, 1);   // next chunk of this W' row
      stage16(&B_s[row][col], src);
    }
    stage_fence();

    const unsigned short* arow = &A_s[mt * 16 + ln][0];
    const unsigned short* brow0 = &B_s[g0 * 16 + ln][0];
    const unsigned short* brow1 = &B_s[(g0 + 1) * 16 + ln][0];
    const int ksteps = len >> 5;         // 4 or 1
    #pragma unroll
    for (int kk = 0; kk < 4; ++kk) {
      if (kk >= ksteps) break;
      int k0 = kk << 5;
      v16bf a = load16(arow, k0, half);
      v16bf b0 = load16(brow0, k0, half);
      v16bf b1 = load16(brow1, k0, half);
      acc0 = __builtin_amdgcn_wmma_f32_16x16x32_bf16(false, a, false, b0,
                                                     (short)0, acc0, false, false);
      acc1 = __builtin_amdgcn_wmma_f32_16x16x32_bf16(false, a, false, b1,
                                                     (short)0, acc1, false, false);
    }
    __syncthreads();                     // LDS reuse across chunks
  }

  // bias + park gate tiles (C layout: VGPR r -> M = r + 8*half, N = ln)
  const float bv0 = bias[g0 * H_SZ + nt * 16 + ln];
  const float bv1 = bias[(g0 + 1) * H_SZ + nt * 16 + ln];
  #pragma unroll
  for (int r = 0; r < 8; ++r) {
    gtile[mt][g0][r + (half << 3)][ln]     = acc0[r] + bv0;
    gtile[mt][g0 + 1][r + (half << 3)][ln] = acc1[r] + bv1;
  }
  __syncthreads();

  // fused LSTM cell: 128 rows x 16 cols = 2048 cells, 4 per thread
  #pragma unroll
  for (int q = 0; q < 4; ++q) {
    int idx = tid * 4 + q;
    int mg = idx >> 4, n = idx & 15;     // mg: batch row 0..127
    int mtt = mg >> 4, m = mg & 15;
    float iv = gtile[mtt][0][m][n], fv = gtile[mtt][1][m][n];
    float gv = gtile[mtt][2][m][n], ov = gtile[mtt][3][m][n];
    long long ci = (long long)mg * H_SZ + nt * 16 + n;
    float cp = cstate[ci];
    float cn = sigmf(fv) * cp + sigmf(iv) * tanhf(gv);
    cstate[ci] = cn;
    float hv = sigmf(ov) * tanhf(cn);
    hout[ci] = f2bf(hv);
  }
}

// ---------- head: out = hs@Wd^T+bd ; logits = out@Wo^T+bo ; log_softmax ----------
__global__ __launch_bounds__(256)
void head_kernel(const unsigned short* __restrict__ hs,    // [T*B, 1024] bf16
                 const unsigned short* __restrict__ Wd16,  // [512,1024] bf16
                 const float* __restrict__ bd,
                 const unsigned short* __restrict__ Wo16,  // [256,512] bf16
                 const float* __restrict__ bo,
                 float* __restrict__ out) {                // [B,T,V] f32
  __shared__ __align__(16) unsigned short out_s[16][D_SZ]; // bf16 intermediate
  __shared__ __align__(16) float logit_s[16][V_SZ];
  __shared__ float lz[16];

  const int tid  = threadIdx.x;
  const int wave = tid >> 5;
  const int lane = tid & 31;
  const int half = lane >> 4;
  const int ln   = lane & 15;
  const int gm0  = blockIdx.x * 16;                        // first global row

  const unsigned short* arow = hs + (long long)(gm0 + ln) * H_SZ;

  // stage 1: out[16,512], wave w covers cols [64w, 64w+64)
  #pragma unroll
  for (int s = 0; s < 4; ++s) {
    int col = (wave * 4 + s) * 16 + ln;
    const unsigned short* brow = Wd16 + (long long)col * H_SZ;
    v8f acc = {0.f, 0.f, 0.f, 0.f, 0.f, 0.f, 0.f, 0.f};
    for (int kk = 0; kk < 32; ++kk) {
      int k0 = kk << 5;
      v16bf a = load16(arow, k0, half);
      v16bf b = load16(brow, k0, half);
      acc = __builtin_amdgcn_wmma_f32_16x16x32_bf16(false, a, false, b,
                                                    (short)0, acc, false, false);
    }
    float bv = bd[col];
    #pragma unroll
    for (int r = 0; r < 8; ++r)
      out_s[r + (half << 3)][col] = f2bf(acc[r] + bv);
  }
  __syncthreads();

  // stage 2: logits[16,256] from LDS, wave w covers cols [32w, 32w+32)
  const unsigned short* arow2 = &out_s[ln][0];
  #pragma unroll
  for (int s = 0; s < 2; ++s) {
    int col = (wave * 2 + s) * 16 + ln;
    const unsigned short* brow = Wo16 + (long long)col * D_SZ;
    v8f acc = {0.f, 0.f, 0.f, 0.f, 0.f, 0.f, 0.f, 0.f};
    #pragma unroll 4
    for (int kk = 0; kk < 16; ++kk) {
      int k0 = kk << 5;
      v16bf a = load16(arow2, k0, half);
      v16bf b = load16(brow, k0, half);
      acc = __builtin_amdgcn_wmma_f32_16x16x32_bf16(false, a, false, b,
                                                    (short)0, acc, false, false);
    }
    float bv = bo[col];
    #pragma unroll
    for (int r = 0; r < 8; ++r)
      logit_s[r + (half << 3)][col] = acc[r] + bv;
  }
  __syncthreads();

  // stage 3: log-softmax per row
  if (tid < 16) {
    float mx = -1e30f;
    for (int v = 0; v < V_SZ; ++v) mx = fmaxf(mx, logit_s[tid][v]);
    float s = 0.0f;
    for (int v = 0; v < V_SZ; ++v) s += __expf(logit_s[tid][v] - mx);
    lz[tid] = mx + __logf(s);
  }
  __syncthreads();

  // transposed store: [T,B] row -> out[b*T*V + t*V + v]
  int r = tid >> 4, q = tid & 15;
  int gm = gm0 + r;
  int t = gm >> 7, b = gm & 127;
  float lzr = lz[r];
  float* orow = out + ((long long)b * T_SEQ + t) * V_SZ + q * 16;
  #pragma unroll
  for (int k = 0; k < 16; ++k)
    orow[k] = logit_s[r][q * 16 + k] - lzr;
}

// ---------- launcher ----------
extern "C" void kernel_launch(void* const* d_in, const int* in_sizes, int n_in,
                              void* d_out, int out_size, void* d_ws, size_t ws_size,
                              hipStream_t stream) {
  const int*   x        = (const int*)  d_in[0];
  const float* emb      = (const float*)d_in[1];
  const float* lstm_Wih = (const float*)d_in[2];
  const float* lstm_Whh = (const float*)d_in[3];
  const float* lstm_bih = (const float*)d_in[4];
  const float* lstm_bhh = (const float*)d_in[5];
  const float* Wd       = (const float*)d_in[6];
  const float* bd       = (const float*)d_in[7];
  const float* Wo       = (const float*)d_in[8];
  const float* bo       = (const float*)d_in[9];
  const float* s1_Wih   = (const float*)d_in[10];
  const float* s1_bih   = (const float*)d_in[11];
  const float* s1_bhh   = (const float*)d_in[12];
  const float* s1_Wd    = (const float*)d_in[13];
  const float* s1_bd    = (const float*)d_in[14];
  const float* s2_Wih   = (const float*)d_in[15];
  const float* s2_bih   = (const float*)d_in[16];
  const float* s2_bhh   = (const float*)d_in[17];
  const float* s2_Wd    = (const float*)d_in[18];
  const float* s2_bd    = (const float*)d_in[19];
  float* outp = (float*)d_out;

  // workspace carve-up (256B aligned slabs)
  char* ws = (char*)d_ws;
  size_t off = 0;
  auto alloc = [&](size_t bytes) {
    char* p = ws + off;
    off += (bytes + 255) & ~(size_t)255;
    return p;
  };
  unsigned short* Wp     = (unsigned short*)alloc((size_t)G4H * KTOT * 2);
  unsigned short* Wd16   = (unsigned short*)alloc((size_t)D_SZ * H_SZ * 2);
  unsigned short* Wo16   = (unsigned short*)alloc((size_t)V_SZ * D_SZ * 2);
  float*          bias   = (float*)         alloc((size_t)G4H * 4);
  float*          cstate = (float*)         alloc((size_t)B_SZ * H_SZ * 4);
  unsigned short* h0     = (unsigned short*)alloc((size_t)B_SZ * H_SZ * 2);
  unsigned short* xin16  = (unsigned short*)alloc((size_t)T_SEQ * B_SZ * XPAD * 2);
  unsigned short* hs16   = (unsigned short*)alloc((size_t)T_SEQ * B_SZ * H_SZ * 2);
  (void)ws_size; (void)in_sizes; (void)n_in; (void)out_size;

  // 1) weight packing / init
  pack_wp_kernel<<<2048, 256, 0, stream>>>(lstm_Whh, lstm_Wih, Wp);
  cast_bf16_kernel<<<1024, 256, 0, stream>>>(Wd, Wd16, D_SZ * H_SZ);
  cast_bf16_kernel<<<256, 256, 0, stream>>>(Wo, Wo16, V_SZ * D_SZ);
  prep_misc_kernel<<<(B_SZ * H_SZ) / 256, 256, 0, stream>>>(
      lstm_bih, lstm_bhh, bias, cstate, h0);

  // 2) embedding + senti attention for all tokens
  embed_senti_kernel<<<(T_SEQ * B_SZ) / 256, 256, 0, stream>>>(
      x, emb, s1_Wih, s1_bih, s1_bhh, s1_Wd, s1_bd,
      s2_Wih, s2_bih, s2_bhh, s2_Wd, s2_bd, xin16);

  // 3) sequential LSTM over T; W' (~9.7MB bf16) read once per step from L2
  for (int t = 0; t < T_SEQ; ++t) {
    const unsigned short* hprev =
        (t == 0) ? h0 : hs16 + (size_t)(t - 1) * B_SZ * H_SZ;
    lstm_step_kernel<<<H_SZ / 16, 512, 0, stream>>>(
        hprev, xin16 + (size_t)t * B_SZ * XPAD, Wp, bias, cstate,
        hs16 + (size_t)t * B_SZ * H_SZ);
  }

  // 4) dense head + log-softmax, transposed store to [B,T,V]
  head_kernel<<<(T_SEQ * B_SZ) / 16, 256, 0, stream>>>(
      hs16, Wd16, bd, Wo16, bo, outp);
}
